// GraphUnet_88364657147967
// MI455X (gfx1250) — matrix operand order repaired and checked by reading, not compiled, not run on hardware
//
#include <hip/hip_runtime.h>
#include <hip/hip_bf16.h>

typedef __attribute__((ext_vector_type(16))) _Float16 v16h;
typedef __attribute__((ext_vector_type(8)))  _Float16 v8h;
typedef __attribute__((ext_vector_type(8)))  float    v8f;

#define NNODES 8192
#define KSPLIT 4

// ---------------------------------------------------------------------------
// Convert dense adjacency f32 -> f16 (entries are 0/1 -> exact). 128MB result
// is L2-resident (192MB L2) for the later SpMM passes.
__global__ void cvt_k(const float* __restrict__ A, _Float16* __restrict__ AH) {
  size_t i = ((size_t)blockIdx.x * blockDim.x + threadIdx.x) * 4;
  float4 f = *reinterpret_cast<const float4*>(A + i);
  union { _Float16 h[4]; uint2 u; } pk;
  pk.h[0] = (_Float16)f.x; pk.h[1] = (_Float16)f.y;
  pk.h[2] = (_Float16)f.z; pk.h[3] = (_Float16)f.w;
  *reinterpret_cast<uint2*>(AH + i) = pk.u;
}

// ---------------------------------------------------------------------------
// Dense feature GEMM + pack transposed f16 with aux column:
//   outT[col][row] = col<NF  : m*(s*(in[row,:]@W[:,col]) + b[col])
//                    col==NF : m            (aux: ones / pool mask -> rowsums)
//                    col>NF  : 0
__global__ void gemm_pack_k(const float* __restrict__ in, int K,
                            const float* __restrict__ W,
                            const float* __restrict__ bias,
                            const float* __restrict__ marr,
                            const float* __restrict__ sarr,
                            _Float16* __restrict__ outT, int NF, int NCOL) {
  int gid = blockIdx.x * blockDim.x + threadIdx.x;
  int col = gid >> 13;
  int row = gid & (NNODES - 1);
  if (col >= NCOL) return;
  float m = marr ? marr[row] : 1.0f;
  float val;
  if (col < NF) {
    float s = sarr ? sarr[row] : 1.0f;
    float acc = 0.0f;
    const float* ip = in + (size_t)row * K;
    for (int k = 0; k < K; ++k) acc = fmaf(ip[k], W[(size_t)k * NF + col], acc);
    val = m * (s * acc + bias[col]);
  } else if (col == NF) {
    val = m;
  } else {
    val = 0.0f;
  }
  outT[(size_t)col * NNODES + row] = (_Float16)val;
}

// ---------------------------------------------------------------------------
// SpMM partial: Yp[ks][row][col] += A[row, kslice] @ X[kslice, col]
// via v_wmma_f32_16x16x32_f16. Block = 64 thr (2 waves), 32 rows, grid.y = K split.
template <typename AT, int NCOL>
__global__ __launch_bounds__(64) void spmm_part_k(const AT* __restrict__ A,
                                                  const _Float16* __restrict__ XT,
                                                  float* __restrict__ Yp) {
  constexpr int NT = NCOL / 16;
  __shared__ alignas(64) _Float16 As[32][32];     // A tile, f16
  __shared__ alignas(64) _Float16 XsT[NCOL][32];  // X^T tile (B operand)

  const int tid  = threadIdx.x;
  const int wave = tid >> 5;
  const int lane = tid & 31;
  const int hh   = lane >> 4;   // lane half
  const int nn   = lane & 15;   // M (for A) / N (for B,D) index
  const int rowblock = blockIdx.x * 32;
  const int ks   = blockIdx.y;
  const int kbeg = ks * (NNODES / KSPLIT);

  v8f acc[NT];
  v8f zero = {0.f, 0.f, 0.f, 0.f, 0.f, 0.f, 0.f, 0.f};
#pragma unroll
  for (int t = 0; t < NT; ++t) acc[t] = zero;

  const int arr   = tid >> 1;   // staged A row 0..31
  const int ahalf = tid & 1;    // which 16-col half
  const AT* aptr = A + (size_t)(rowblock + arr) * NNODES + ahalf * 16;

  for (int kk = 0; kk < NNODES / KSPLIT; kk += 32) {
    const int k0 = kbeg + kk;
    __syncthreads();
    // ---- stage A tile (32x32) into LDS as f16
    if constexpr (sizeof(AT) == 4) {
      const float4* s4 = reinterpret_cast<const float4*>(aptr + k0);
      float4 f0 = s4[0], f1 = s4[1], f2 = s4[2], f3 = s4[3];
      v8h lo, hi;
      lo[0]=(_Float16)f0.x; lo[1]=(_Float16)f0.y; lo[2]=(_Float16)f0.z; lo[3]=(_Float16)f0.w;
      lo[4]=(_Float16)f1.x; lo[5]=(_Float16)f1.y; lo[6]=(_Float16)f1.z; lo[7]=(_Float16)f1.w;
      hi[0]=(_Float16)f2.x; hi[1]=(_Float16)f2.y; hi[2]=(_Float16)f2.z; hi[3]=(_Float16)f2.w;
      hi[4]=(_Float16)f3.x; hi[5]=(_Float16)f3.y; hi[6]=(_Float16)f3.z; hi[7]=(_Float16)f3.w;
      *reinterpret_cast<v8h*>(&As[arr][ahalf * 16])     = lo;
      *reinterpret_cast<v8h*>(&As[arr][ahalf * 16 + 8]) = hi;
    } else {
      const uint4* s4 = reinterpret_cast<const uint4*>(aptr + k0);
      uint4* d = reinterpret_cast<uint4*>(&As[arr][ahalf * 16]);
      d[0] = s4[0]; d[1] = s4[1];
    }
    // ---- stage X^T tile: NCOL rows x 32 K, contiguous 64B per row
    for (int c = tid; c < NCOL; c += 64) {
      const uint4* sp = reinterpret_cast<const uint4*>(XT + (size_t)c * NNODES + k0);
      uint4* dp = reinterpret_cast<uint4*>(&XsT[c][0]);
      dp[0] = sp[0]; dp[1] = sp[1]; dp[2] = sp[2]; dp[3] = sp[3];
    }
    if (kk + 32 < NNODES / KSPLIT) __builtin_prefetch(aptr + k0 + 32, 0, 1);
    __syncthreads();

    // ---- A fragment: lane nn holds row wave*16+nn; K = 16h..16h+15 contiguous
    v16h a;
    {
      const int am = wave * 16 + nn;
      a = *reinterpret_cast<const v16h*>(&As[am][16 * hh]);
    }
#pragma unroll
    for (int nt = 0; nt < NT; ++nt) {
      // B fragment: lane nn = col, K = 16h..16h+15 contiguous in XsT row
      v16h b = *reinterpret_cast<const v16h*>(&XsT[nt * 16 + nn][16 * hh]);
      acc[nt] = __builtin_amdgcn_wmma_f32_16x16x32_f16(false, a, false, b,
                                                       (short)0, acc[nt],
                                                       false, false);
    }
  }

  // ---- write partial accumulators (D: row = r + 8*hh, col = nt*16 + nn)
#pragma unroll
  for (int nt = 0; nt < NT; ++nt) {
#pragma unroll
    for (int r = 0; r < 8; ++r) {
      const int rg  = rowblock + wave * 16 + r + 8 * hh;
      const int col = nt * 16 + nn;
      Yp[((size_t)ks * NNODES + rg) * NCOL + col] = acc[nt][r];
    }
  }
}

// ---------------------------------------------------------------------------
// Reduce K partials + gconv epilogue: h = act( (Y + t_self) / (rowsum+1) ) * mask
template <int NCOL, int NF, bool RELU, bool MASK>
__global__ void reduce_k(const float* __restrict__ Yp,
                         const _Float16* __restrict__ XT,
                         const float* __restrict__ Mrow,
                         float* __restrict__ Hout) {
  int gid = blockIdx.x * blockDim.x + threadIdx.x;
  int rg  = gid / NF;
  int col = gid - rg * NF;
  float y = 0.f, rsum = 0.f;
#pragma unroll
  for (int ks = 0; ks < KSPLIT; ++ks) {
    const float* base = Yp + ((size_t)ks * NNODES + rg) * NCOL;
    y    += base[col];
    rsum += base[NF];          // aux column = A @ (ones|mask) = row degree
  }
  y = (y + (float)XT[(size_t)col * NNODES + rg]) / (rsum + 1.0f); // +I self loop
  if (RELU) y = fmaxf(y, 0.0f);
  if (MASK) y *= Mrow[rg];
  Hout[(size_t)rg * NF + col] = y;
}

// ---------------------------------------------------------------------------
// Raw pool scores: sraw[row] = h0[row,:] @ pw + pb   (one wave per row)
__global__ void score_k(const float* __restrict__ Hmat, const float* __restrict__ pw,
                        const float* __restrict__ pb, float* __restrict__ sraw) {
  const int wave = threadIdx.x >> 5;
  const int lane = threadIdx.x & 31;
  const int row  = blockIdx.x * 8 + wave;
  const float4 hv = *reinterpret_cast<const float4*>(Hmat + (size_t)row * 128 + lane * 4);
  const float4 wv = *reinterpret_cast<const float4*>(pw + lane * 4);
  float d = hv.x * wv.x + hv.y * wv.y + hv.z * wv.z + hv.w * wv.w;
  for (int off = 16; off > 0; off >>= 1) d += __shfl_down(d, off, 32);
  if (lane == 0) sraw[row] = d + pb[0];
}

// ---------------------------------------------------------------------------
// L2-normalize -> sigmoid -> exact top-4096 via single-block LDS bitonic sort.
// Emits per-row mask (0/1) and gate value (sigmoid score if selected).
__global__ __launch_bounds__(1024) void topk_k(const float* __restrict__ sraw,
                                               float* __restrict__ rowm,
                                               float* __restrict__ rows) {
  __shared__ float key[NNODES];
  __shared__ unsigned short gi[NNODES];
  __shared__ float red[1024];
  const int tid = threadIdx.x;
  float ssq = 0.f;
  for (int g = tid; g < NNODES; g += 1024) {
    float v = sraw[g];
    key[g] = v; gi[g] = (unsigned short)g;
    ssq += v * v;
  }
  red[tid] = ssq; __syncthreads();
  for (int s = 512; s > 0; s >>= 1) {
    if (tid < s) red[tid] += red[tid + s];
    __syncthreads();
  }
  const float rn = rsqrtf(red[0]);
  for (int g = tid; g < NNODES; g += 1024)
    key[g] = 1.0f / (1.0f + expf(-key[g] * rn));
  __syncthreads();
  // bitonic sort, descending
  for (unsigned size = 2; size <= NNODES; size <<= 1) {
    for (unsigned str = size >> 1; str > 0; str >>= 1) {
      for (unsigned w = tid; w < NNODES / 2; w += 1024) {
        unsigned i = 2 * w - (w & (str - 1));
        unsigned j = i + str;
        bool dirDesc = (i & size) == 0;
        float ki = key[i], kj = key[j];
        if ((ki < kj) == dirDesc) {
          key[i] = kj; key[j] = ki;
          unsigned short t = gi[i]; gi[i] = gi[j]; gi[j] = t;
        }
      }
      __syncthreads();
    }
  }
  for (int p = tid; p < NNODES; p += 1024) {
    int g = gi[p];
    bool sel = p < NNODES / 2;   // k = 4096
    rowm[g] = sel ? 1.0f : 0.0f;
    rows[g] = sel ? key[p] : 0.0f;
  }
}

// ---------------------------------------------------------------------------
__global__ void lsm_k(const float* __restrict__ hf, float* __restrict__ out) {
  const int row = blockIdx.x * blockDim.x + threadIdx.x;
  const float* v = hf + (size_t)row * 16;
  float m = v[0];
#pragma unroll
  for (int c = 1; c < 16; ++c) m = fmaxf(m, v[c]);
  float s = 0.f;
#pragma unroll
  for (int c = 0; c < 16; ++c) s += expf(v[c] - m);
  const float l = logf(s) + m;
#pragma unroll
  for (int c = 0; c < 16; ++c) out[(size_t)row * 16 + c] = v[c] - l;
}

// ---------------------------------------------------------------------------
extern "C" void kernel_launch(void* const* d_in, const int* in_sizes, int n_in,
                              void* d_out, int out_size, void* d_ws, size_t ws_size,
                              hipStream_t stream) {
  (void)in_sizes; (void)n_in; (void)out_size;
  const float* x  = (const float*)d_in[0];
  const float* A  = (const float*)d_in[1];
  const float* W0 = (const float*)d_in[2];
  const float* b0 = (const float*)d_in[3];
  const float* W1 = (const float*)d_in[4];
  const float* b1 = (const float*)d_in[5];
  const float* pw = (const float*)d_in[6];
  const float* pb = (const float*)d_in[7];
  const float* Wu = (const float*)d_in[8];
  const float* bu = (const float*)d_in[9];
  const float* Wf = (const float*)d_in[10];
  const float* bf = (const float*)d_in[11];
  float* out = (float*)d_out;

  uint8_t* p = (uint8_t*)d_ws;
  _Float16* XT   = (_Float16*)p;                          // 144*8192*2  = 2359296
  float*    Hbuf = (float*)(p + 2359296);                 // 8192*128*4  = 4194304
  float*    HF   = (float*)(p + 6553600);                 // 8192*16*4   = 524288
  float*    SRAW = (float*)(p + 7077888);                 // 32768
  float*    ROWM = (float*)(p + 7110656);                 // 32768
  float*    ROWS = (float*)(p + 7143424);                 // 32768
  float*    YP   = (float*)(p + 7176192);                 // 4*8192*144*4 = 18874368
  const size_t offAH = 7176192 + 18874368;                // = 26050560
  const size_t ahBytes = (size_t)NNODES * NNODES * 2;     // 128MB
  bool useAH = ws_size >= offAH + ahBytes;
  _Float16* AH = (_Float16*)(p + offAH);

  if (useAH) cvt_k<<<65536, 256, 0, stream>>>(A, AH);     // A -> f16 (L2-resident)

  // 1) t0 = x@W0 + b0 (aux = ones)
  gemm_pack_k<<<(NNODES * 144) / 256, 256, 0, stream>>>(x, 256, W0, b0,
                                                        nullptr, nullptr, XT, 128, 144);
  // 2) h0 = relu(D^-1 (A+I) t0)
  if (useAH) spmm_part_k<_Float16, 144><<<dim3(256, KSPLIT), 64, 0, stream>>>(AH, XT, YP);
  else       spmm_part_k<float,    144><<<dim3(256, KSPLIT), 64, 0, stream>>>(A,  XT, YP);
  reduce_k<144, 128, true, false><<<(NNODES * 128) / 256, 256, 0, stream>>>(YP, XT, nullptr, Hbuf);
  // 3-4) pooling scores + exact top-k -> mask/gate
  score_k<<<NNODES / 8, 256, 0, stream>>>(Hbuf, pw, pb, SRAW);
  topk_k<<<1, 1024, 0, stream>>>(SRAW, ROWM, ROWS);
  // 5) t1_full = mask*(gate*h0 @ W1 + b1) (aux = mask -> pooled rowsums)
  gemm_pack_k<<<(NNODES * 144) / 256, 256, 0, stream>>>(Hbuf, 128, W1, b1,
                                                        ROWM, ROWS, XT, 128, 144);
  // 6) xu = mask * relu(Dp^-1 (Ap+I) t1)  (pooled gconv on full domain + unpool)
  if (useAH) spmm_part_k<_Float16, 144><<<dim3(256, KSPLIT), 64, 0, stream>>>(AH, XT, YP);
  else       spmm_part_k<float,    144><<<dim3(256, KSPLIT), 64, 0, stream>>>(A,  XT, YP);
  reduce_k<144, 128, true, true><<<(NNODES * 128) / 256, 256, 0, stream>>>(YP, XT, ROWM, Hbuf);
  // 7) tu = xu@Wu + bu
  gemm_pack_k<<<(NNODES * 144) / 256, 256, 0, stream>>>(Hbuf, 128, Wu, bu,
                                                        nullptr, nullptr, XT, 128, 144);
  // 8) hu = relu(D^-1 (A+I) tu)
  if (useAH) spmm_part_k<_Float16, 144><<<dim3(256, KSPLIT), 64, 0, stream>>>(AH, XT, YP);
  else       spmm_part_k<float,    144><<<dim3(256, KSPLIT), 64, 0, stream>>>(A,  XT, YP);
  reduce_k<144, 128, true, false><<<(NNODES * 128) / 256, 256, 0, stream>>>(YP, XT, nullptr, Hbuf);
  // 9) tf = hu@Wf + bf (NF=16, NCOL=32)
  gemm_pack_k<<<(NNODES * 32) / 256, 256, 0, stream>>>(Hbuf, 128, Wf, bf,
                                                       nullptr, nullptr, XT, 16, 32);
  // 10) hf = D^-1 (A+I) tf
  if (useAH) spmm_part_k<_Float16, 32><<<dim3(256, KSPLIT), 64, 0, stream>>>(AH, XT, YP);
  else       spmm_part_k<float,    32><<<dim3(256, KSPLIT), 64, 0, stream>>>(A,  XT, YP);
  reduce_k<32, 16, false, false><<<(NNODES * 16) / 256, 256, 0, stream>>>(YP, XT, nullptr, HF);
  // 11) log_softmax
  lsm_k<<<NNODES / 256, 256, 0, stream>>>(HF, out);
}
